// MixtralSparseMoeBlock_61667140436567
// MI455X (gfx1250) — matrix operand order repaired and checked by reading, not compiled, not run on hardware
//
#include <hip/hip_runtime.h>
#include <hip/hip_bf16.h>
#include <cstdint>
#include <cstddef>

typedef __attribute__((ext_vector_type(16))) _Float16 v16h;
typedef __attribute__((ext_vector_type(8)))  float    v8f;

constexpr int T  = 8192;   // BATCH * SEQ
constexpr int D  = 1024;   // HIDDEN
constexpr int F  = 3584;   // FFN
constexpr int E  = 8;      // experts
constexpr int MT = 16;     // token tile (WMMA M)
constexpr int KC = 32;     // WMMA K for f16

// ---------------------------------------------------------------------------
// Router: raw logits -> d_out tail, softmax/top2/renorm -> combine[T,E] in ws,
// plus one-pass f32->f16 conversion of x into ws.
// ---------------------------------------------------------------------------
__global__ void __launch_bounds__(256) router_kernel(
    const float* __restrict__ x, const float* __restrict__ gate_w,
    float* __restrict__ logits, float* __restrict__ combine,
    _Float16* __restrict__ xh) {
  __shared__ float gw[E * D];                       // 32 KB gate weights
  for (int i = threadIdx.x; i < E * D; i += 256) gw[i] = gate_w[i];
  __syncthreads();

  const int t = blockIdx.x * 256 + threadIdx.x;     // one token per thread
  const float* xrow = x + (size_t)t * D;
  _Float16* hrow = xh + (size_t)t * D;

  float acc[E];
#pragma unroll
  for (int e = 0; e < E; ++e) acc[e] = 0.f;

  for (int d = 0; d < D; ++d) {
    float xv = xrow[d];
    hrow[d] = (_Float16)xv;                         // f16 activation copy
#pragma unroll
    for (int e = 0; e < E; ++e) acc[e] += xv * gw[e * D + d];
  }

#pragma unroll
  for (int e = 0; e < E; ++e) logits[(size_t)t * E + e] = acc[e];

  // softmax over 8
  float mx = acc[0];
#pragma unroll
  for (int e = 1; e < E; ++e) mx = fmaxf(mx, acc[e]);
  float p[E], s = 0.f;
#pragma unroll
  for (int e = 0; e < E; ++e) { p[e] = __expf(acc[e] - mx); s += p[e]; }
  float inv = 1.f / s;
#pragma unroll
  for (int e = 0; e < E; ++e) p[e] *= inv;

  // top-2 (first index wins ties, matching lax.top_k)
  int i1 = 0;
#pragma unroll
  for (int e = 1; e < E; ++e) if (p[e] > p[i1]) i1 = e;
  int i2 = (i1 == 0) ? 1 : 0;
#pragma unroll
  for (int e = 0; e < E; ++e) if (e != i1 && p[e] > p[i2]) i2 = e;
  float rs = 1.f / (p[i1] + p[i2]);
#pragma unroll
  for (int e = 0; e < E; ++e)
    combine[(size_t)t * E + e] = (e == i1) ? p[i1] * rs
                               : (e == i2) ? p[i2] * rs : 0.f;
}

// ---------------------------------------------------------------------------
// Fused expert MLP: each block owns a 16-token tile, loops all experts.
// Phase A  : ALL 8 waves compute x@w1^T / x@w3^T for a 32-wide FFN chunk,
//            K=1024 split in half across wave pairs (partial sums in LDS).
// Phase B1 : reduce K-halves, silu(g1)*g3 * combine_weight -> f16 act tile.
// Phase B2 : all 8 waves accumulate act @ w2^T into per-wave [16 x 128] f32
//            register accumulators (8 x v8f) held across the FFN sweep.
// ---------------------------------------------------------------------------
__global__ void __launch_bounds__(256) moe_expert_kernel(
    const _Float16* __restrict__ xh, const float* __restrict__ combine,
    const float* __restrict__ w1, const float* __restrict__ w3,
    const float* __restrict__ w2, float* __restrict__ out) {
  __shared__ _Float16 xs[MT * D];         // 32 KB  x tile, f16, [t][d]
  __shared__ float    g1s[2][MT * KC];    //  4 KB  x@w1^T K-half partials
  __shared__ float    g3s[2][MT * KC];    //  4 KB  x@w3^T K-half partials
  __shared__ _Float16 acts[MT * KC];      //  1 KB  swiglu result (A of GEMM2)
  __shared__ float    cws[MT * E];        //  combine weights for tile

  const int tile0 = blockIdx.x * MT;
  const int tid   = threadIdx.x;
  const int wv    = tid >> 5;             // wave 0..7
  const int lane  = tid & 31;
  const int hi    = (lane >= 16) ? 1 : 0;
  const int nn    = lane & 15;

  // ---- stage x tile via CDNA5 async global->LDS copy (ASYNCcnt-tracked)
  {
    const uint64_t gsrc = (uint64_t)(uintptr_t)(xh + (size_t)tile0 * D);
    const uint32_t lbase = (uint32_t)(uintptr_t)xs;  // LDS offset (aperture low32)
    for (int i = tid; i < (MT * D * 2) / 16; i += 256) {
      uint64_t ga = gsrc + (uint64_t)i * 16u;
      uint32_t la = lbase + (uint32_t)i * 16u;
      asm volatile("global_load_async_to_lds_b128 %0, %1, off"
                   :: "v"(la), "v"(ga) : "memory");
    }
    if (tid < MT * E)
      cws[tid] = combine[(size_t)(tile0 + tid / E) * E + (tid % E)];
    asm volatile("s_wait_asynccnt 0" ::: "memory");
  }
  __syncthreads();

  v8f zero = {};
  v8f acc[8];
#pragma unroll
  for (int j = 0; j < 8; ++j) acc[j] = zero;

  for (int e = 0; e < E; ++e) {
    // uniform skip if no token in this tile routed to expert e
    bool active = false;
#pragma unroll
    for (int m = 0; m < MT; ++m) active |= (cws[m * E + e] != 0.f);
    if (!active) continue;

    for (int f0 = 0; f0 < F; f0 += KC) {
      // ---- Phase A: 16x32 chunk of x@{w1,w3}^T, K split over wave pairs
      {
        const int which = wv & 1;            // 0 -> w1, 1 -> w3
        const int nt    = (wv >> 1) & 1;     // 16-wide ffn subtile
        const int kh    = (wv >> 2) & 1;     // K half: d in [kh*512, kh*512+512)
        const float* wsrc =
            (which ? w3 : w1) + ((size_t)e * F + f0 + nt * 16 + nn) * D;
        __builtin_prefetch(wsrc + KC * D, 0, 1);   // next chunk's rows
        v8f g = zero;
        const int kc0 = kh * (D / KC / 2);
        for (int kc = kc0; kc < kc0 + D / KC / 2; ++kc) {
          const int kb = kc * KC;
          v16h a, b;
#pragma unroll
          for (int i = 0; i < 16; ++i) {
            // A (16x32 f16): lanes 0-15 hold K 0..15; 16-31 hold K 16..31
            int ka = hi * 16 + i;
            a[i] = xs[nn * D + kb + ka];
            // B (32x16 f16): lane = column (ffn row), contiguous K per lane
            b[i] = (_Float16)wsrc[kb + hi * 16 + i];
          }
          g = __builtin_amdgcn_wmma_f32_16x16x32_f16(false, a, false, b,
                                                     (short)0, g, false, false);
        }
        float* gdst = which ? g3s[kh] : g1s[kh];
#pragma unroll
        for (int r = 0; r < 8; ++r)
          gdst[(hi * 8 + r) * KC + nt * 16 + nn] = g[r];
      }
      __syncthreads();

      // ---- Phase B1: reduce K halves; act = combine[t]*silu(g1)*g3 (f16)
      for (int i = tid; i < MT * KC; i += 256) {
        float g1  = g1s[0][i] + g1s[1][i];
        float g3  = g3s[0][i] + g3s[1][i];
        float sig = 1.f / (1.f + __expf(-g1));
        acts[i] = (_Float16)(cws[(i >> 5) * E + e] * (g1 * sig * g3));
      }
      __syncthreads();

      // ---- Phase B2: acc += act @ w2[:,f0:f0+32]^T  (wave covers 128 dims)
      v16h a;
#pragma unroll
      for (int i = 0; i < 16; ++i) {
        int ka = hi * 16 + i;
        a[i] = acts[nn * KC + ka];
      }
#pragma unroll
      for (int j = 0; j < 8; ++j) {
        const float* w2row =
            w2 + ((size_t)e * D + wv * 128 + j * 16 + nn) * F + f0;
        __builtin_prefetch(w2row + KC, 0, 1);
        v16h b;
#pragma unroll
        for (int i = 0; i < 16; ++i) b[i] = (_Float16)w2row[hi * 16 + i];
        acc[j] = __builtin_amdgcn_wmma_f32_16x16x32_f16(false, a, false, b,
                                                        (short)0, acc[j],
                                                        false, false);
      }
      // (no barrier needed here: next Phase A writes only g1s/g3s, reads xs)
    }
  }

  // ---- store: block covers its 16 rows completely, exactly once
#pragma unroll
  for (int j = 0; j < 8; ++j) {
#pragma unroll
    for (int r = 0; r < 8; ++r) {
      out[(size_t)(tile0 + hi * 8 + r) * D + wv * 128 + j * 16 + nn] =
          acc[j][r];
    }
  }
}

// ---------------------------------------------------------------------------
extern "C" void kernel_launch(void* const* d_in, const int* in_sizes, int n_in,
                              void* d_out, int out_size, void* d_ws,
                              size_t ws_size, hipStream_t stream) {
  const float* x      = (const float*)d_in[0];  // [B,S,D] -> [T,D]
  const float* gate_w = (const float*)d_in[1];  // [E,D]
  const float* w1     = (const float*)d_in[2];  // [E,F,D]
  const float* w3     = (const float*)d_in[3];  // [E,F,D]
  const float* w2     = (const float*)d_in[4];  // [E,D,F]

  float* out    = (float*)d_out;                // [T,D]
  float* logits = out + (size_t)T * D;          // [T,E] tail of d_out

  float*    combine = (float*)d_ws;                              // T*E f32
  _Float16* xh = (_Float16*)((char*)d_ws + (size_t)T * E * 4);   // T*D f16

  router_kernel<<<T / 256, 256, 0, stream>>>(x, gate_w, logits, combine, xh);
  moe_expert_kernel<<<T / MT, 256, 0, stream>>>(xh, combine, w1, w3, w2, out);
}
